// SingleLSTM_7541962571854
// MI455X (gfx1250) — matrix operand order repaired and checked by reading, not compiled, not run on hardware
//
#include <hip/hip_runtime.h>
#include <hip/hip_bf16.h>
#include <stdint.h>

// ---------------------------------------------------------------------------
// 2-layer LSTM encoder (8 steps) + autoregressive decoder (12 steps), B=32768,
// HID=256.  bf16 WMMA (v_wmma_f32_16x16x32_bf16) with f32 accumulation.
// Persistent-state design: each block owns 32 batch rows for all 20 timesteps;
// c-state lives in VGPRs, h-state bounces through LDS for the C->A layout
// redistribution.  The K=3 input projection AND the gate biases are folded
// into the GEMMs via a 9th K-tile (x | 1.0 staged in extended h0 columns).
// Weight fragments stream from L2 every step; an opaque (asm-laundered) zero
// offset keeps the loads loop-variant so LICM can't hoist/spill them, while
// preserving global address-space inference (global_load_b128, not flat).
// ---------------------------------------------------------------------------

typedef __attribute__((ext_vector_type(16))) __bf16 v16bf;
typedef __attribute__((ext_vector_type(8)))  __bf16 v8bf;
typedef __attribute__((ext_vector_type(8)))  float  v8f;

#define HID        256
#define GATES      1024          // 4*HID
#define KT9        9             // extended GEMM K tiles: 256/32 + 1
#define KT8        8             // 256 / 32
#define MROWS      32            // batch rows per block (2 M-tiles)
#define WAVES      16            // one wave per 16 hidden columns
#define THREADS    (WAVES * 32)
#define LST0       296           // sH0 row stride in bf16 (288 + 8 pad)
#define LST1       264           // sH1 row stride in bf16 (256 + 8 pad)
#define OBS_LEN    8
#define PRED_LEN   12
#define TOT_STEPS  (OBS_LEN + PRED_LEN)
#define MAT9_ELEMS (64 * KT9 * 32 * 16)   // 294912 (ext: Whh0+Wih0+b0 / Wih1+b1)
#define MAT8_ELEMS (64 * KT8 * 32 * 16)   // 262144 (W_hh1)

__device__ __forceinline__ unsigned short f2bf(float f) {
    unsigned u = __float_as_uint(f);
    u += 0x7FFFu + ((u >> 16) & 1u);        // round-to-nearest-even
    return (unsigned short)(u >> 16);
}
__device__ __forceinline__ float bf2f(unsigned short b) {
    return __uint_as_float(((unsigned)b) << 16);
}
__device__ __forceinline__ float sigf(float x) {
    return __builtin_amdgcn_rcpf(1.0f + __expf(-x));
}
__device__ __forceinline__ float tanh_fast(float x) {
    // tanh(x) = 1 - 2/(exp(2x)+1)   (v_exp_f32 + v_rcp_f32)
    return __builtin_fmaf(-2.0f, __builtin_amdgcn_rcpf(__expf(2.0f * x) + 1.0f), 1.0f);
}

// ---------------------------------------------------------------------------
// Pack weights into bf16 B-fragment order for v_wmma_f32_16x16x32_bf16:
//   per 32(K) x 16(N) tile, lane l holds column N = ntile*16 + (l&15),
//   elements e=0..15 hold K = ktile*32 + e + ((l&16)?16:0)   (contiguous k order; loadA matches)
// Matrix 0 (9 K-tiles): W_hh0; tile 8 rows 0..2 = W_ih0, row 3 = b_ih0+b_hh0.
// Matrix 1 (9 K-tiles): W_ih1; tile 8 row 3 = b_ih1+b_hh1 (rows 0..2 zero so
//                       the staged x does not leak into layer 1).
// Matrix 2 (8 K-tiles): W_hh1.
// ---------------------------------------------------------------------------
__global__ void pack_kernel(const float* __restrict__ Whh0,
                            const float* __restrict__ Wih0,
                            const float* __restrict__ Wih1,
                            const float* __restrict__ Whh1,
                            const float* __restrict__ bih0,
                            const float* __restrict__ bhh0,
                            const float* __restrict__ bih1,
                            const float* __restrict__ bhh1,
                            unsigned short* __restrict__ packed) {
    int idx = blockIdx.x * blockDim.x + threadIdx.x;
    if (idx < 2 * MAT9_ELEMS) {
        int mat  = idx / MAT9_ELEMS;
        int rem  = idx % MAT9_ELEMS;
        int e    = rem & 15;
        int lane = (rem >> 4) & 31;
        int t9   = rem >> 9;
        int kt   = t9 % KT9;
        int nt   = t9 / KT9;
        int n    = nt * 16 + (lane & 15);
        int kk   = e + ((lane & 16) ? 16 : 0);     // 0..31 within tile
        float v;
        if (kt < 8) {
            v = (mat == 0 ? Whh0 : Wih1)[n * HID + kt * 32 + kk];
        } else if (mat == 0) {
            v = (kk < 3) ? Wih0[n * 3 + kk]
              : (kk == 3) ? (bih0[n] + bhh0[n]) : 0.0f;
        } else {
            v = (kk == 3) ? (bih1[n] + bhh1[n]) : 0.0f;
        }
        packed[idx] = f2bf(v);
    } else if (idx < 2 * MAT9_ELEMS + MAT8_ELEMS) {
        int rem  = idx - 2 * MAT9_ELEMS;
        int e    = rem & 15;
        int lane = (rem >> 4) & 31;
        int kt   = (rem >> 9) & 7;
        int nt   = rem >> 12;
        int n    = nt * 16 + (lane & 15);
        int k    = kt * 32 + e + ((lane & 16) ? 16 : 0);
        packed[idx] = f2bf(Whh1[n * HID + k]);
    }
}

// Load one 16x32 bf16 A fragment from LDS (row-major h tile).
// Lane l: row = mbase + (l&15); two 8-half chunks at k offsets klo, klo+8.
__device__ __forceinline__ v16bf loadA(const unsigned short* sH, int stride,
                                       int mbase, int kb, int lane) {
    const int klo = (lane & 16) ? 16 : 0;
    union { v16bf v; v8bf h[2]; } u;
    const unsigned short* p = sH + (mbase + (lane & 15)) * stride + kb + klo;
    u.h[0] = *(const v8bf*)(const void*)(p);
    u.h[1] = *(const v8bf*)(const void*)(p + 8);
    return u.v;
}

__global__ void __launch_bounds__(THREADS, 1)
lstm_kernel(const float* __restrict__ obs,        // [8, B, 3]
            const float* __restrict__ Wout,       // [3, 256]
            const float* __restrict__ bout,       // [3]
            const unsigned short* __restrict__ packed,
            float* __restrict__ out,              // [12, B, 3]
            int B) {
    __shared__ __align__(16) unsigned short sH0[MROWS * LST0]; // h0 | x,1,pad
    __shared__ __align__(16) unsigned short sH1[MROWS * LST1];
    __shared__ float sWout[3 * HID];
    __shared__ float sP[96][4];

    const int tid     = threadIdx.x;
    const int wave    = tid >> 5;
    const int lane    = tid & 31;
    const int col     = (wave << 4) + (lane & 15);    // hidden column 0..255
    const int hi8     = (lane & 16) ? 8 : 0;          // C/D row offset
    const int rowbase = blockIdx.x * MROWS;

    // zero initial h state (also zeroes x pad cols once; they stay zero)
    for (int i = tid; i < MROWS * LST0; i += THREADS) sH0[i] = 0;
    for (int i = tid; i < MROWS * LST1; i += THREADS) sH1[i] = 0;
    for (int i = tid; i < 3 * HID; i += THREADS)      sWout[i] = Wout[i];
    if (tid < MROWS) sH0[tid * LST0 + HID + 3] = 0x3F80; // bf16(1.0): bias lane

    // per-wave loop-invariant fragment bases
    const unsigned short* pb0[4];
    const unsigned short* pb1[4];
    const unsigned short* pb2[4];
#pragma unroll
    for (int g = 0; g < 4; ++g) {
        int nt = g * 16 + wave;
        pb0[g] = packed + (size_t)nt * (KT9 * 512) + (size_t)lane * 16;
        pb1[g] = packed + MAT9_ELEMS + (size_t)nt * (KT9 * 512) + (size_t)lane * 16;
        pb2[g] = packed + 2 * MAT9_ELEMS + (size_t)nt * (KT8 * 512)
                        + (size_t)lane * 16;
    }

    // persistent cell state in VGPRs: c[layer][mtile] -> v8f (rows r+hi8, col)
    v8f c0[2], c1[2];
#pragma unroll
    for (int mt = 0; mt < 2; ++mt) {
        c0[mt] = (v8f){0.f,0.f,0.f,0.f,0.f,0.f,0.f,0.f};
        c1[mt] = (v8f){0.f,0.f,0.f,0.f,0.f,0.f,0.f,0.f};
    }

#pragma unroll 1
    for (int t = 0; t < TOT_STEPS; ++t) {
        // Opaque zero offset: keeps weight-load addresses loop-variant so
        // LICM cannot hoist (and spill) them out of the time loop, while the
        // loads stay based on `packed` (global address space inference).
        int toff = 0;
        asm volatile("" : "+v"(toff));

        // ---- stage x (bf16) into extended h0 columns 256..258 ----
        if (t <= OBS_LEN && tid < MROWS * 3) {   // t==8 reuses obs[7]
            int ts = (t < OBS_LEN) ? t : (OBS_LEN - 1);
            int m = tid / 3, j = tid % 3;
            sH0[m * LST0 + HID + j] =
                f2bf(obs[((size_t)ts * B + rowbase + m) * 3 + j]);
        }
        __syncthreads();        // x ready; prev-step LDS writes visible

        // ========== layer 0: gates = [h0 | x | 1] @ [Whh0 | Wih0 | b0]^T ====
        v8f acc[4][2];
#pragma unroll
        for (int g = 0; g < 4; ++g)
#pragma unroll
            for (int mt = 0; mt < 2; ++mt)
                acc[g][mt] = (v8f){0.f,0.f,0.f,0.f,0.f,0.f,0.f,0.f};
#pragma unroll
        for (int kt = 0; kt < KT9; ++kt) {
            int kb = kt << 5;
            v16bf a0 = loadA(sH0, LST0, 0,  kb, lane);
            v16bf a1 = loadA(sH0, LST0, 16, kb, lane);
#pragma unroll
            for (int g = 0; g < 4; ++g) {
                v16bf bf = *(const v16bf*)(const void*)(pb0[g] + (kt << 9) + toff);
                acc[g][0] = __builtin_amdgcn_wmma_f32_16x16x32_bf16(
                    false, a0, false, bf, (short)0, acc[g][0], false, false);
                acc[g][1] = __builtin_amdgcn_wmma_f32_16x16x32_bf16(
                    false, a1, false, bf, (short)0, acc[g][1], false, false);
            }
        }
        __syncthreads();        // all reads of h0_old complete
#pragma unroll
        for (int mt = 0; mt < 2; ++mt) {
#pragma unroll
            for (int r = 0; r < 8; ++r) {
                float ig = sigf(acc[0][mt][r]);
                float fg = sigf(acc[1][mt][r]);
                float gg = tanh_fast(acc[2][mt][r]);
                float og = sigf(acc[3][mt][r]);
                float c  = fg * c0[mt][r] + ig * gg;
                c0[mt][r] = c;
                sH0[(mt * 16 + r + hi8) * LST0 + col] = f2bf(og * tanh_fast(c));
            }
        }
        __syncthreads();        // h0_new visible

        // ========== layer 1: gates = [h0 | x | 1] @ [Wih1 | 0 | b1]^T
        //                           +  h1_old @ Whh1^T =======================
#pragma unroll
        for (int g = 0; g < 4; ++g)
#pragma unroll
            for (int mt = 0; mt < 2; ++mt)
                acc[g][mt] = (v8f){0.f,0.f,0.f,0.f,0.f,0.f,0.f,0.f};
#pragma unroll
        for (int kt = 0; kt < KT9; ++kt) {
            int kb = kt << 5;
            v16bf a0 = loadA(sH0, LST0, 0,  kb, lane);
            v16bf a1 = loadA(sH0, LST0, 16, kb, lane);
#pragma unroll
            for (int g = 0; g < 4; ++g) {
                v16bf bf = *(const v16bf*)(const void*)(pb1[g] + (kt << 9) + toff);
                acc[g][0] = __builtin_amdgcn_wmma_f32_16x16x32_bf16(
                    false, a0, false, bf, (short)0, acc[g][0], false, false);
                acc[g][1] = __builtin_amdgcn_wmma_f32_16x16x32_bf16(
                    false, a1, false, bf, (short)0, acc[g][1], false, false);
            }
        }
#pragma unroll
        for (int kt = 0; kt < KT8; ++kt) {
            int kb = kt << 5;
            v16bf a0 = loadA(sH1, LST1, 0,  kb, lane);
            v16bf a1 = loadA(sH1, LST1, 16, kb, lane);
#pragma unroll
            for (int g = 0; g < 4; ++g) {
                v16bf bf = *(const v16bf*)(const void*)(pb2[g] + (kt << 9) + toff);
                acc[g][0] = __builtin_amdgcn_wmma_f32_16x16x32_bf16(
                    false, a0, false, bf, (short)0, acc[g][0], false, false);
                acc[g][1] = __builtin_amdgcn_wmma_f32_16x16x32_bf16(
                    false, a1, false, bf, (short)0, acc[g][1], false, false);
            }
        }
        __syncthreads();        // all reads of h1_old complete
#pragma unroll
        for (int mt = 0; mt < 2; ++mt) {
#pragma unroll
            for (int r = 0; r < 8; ++r) {
                float ig = sigf(acc[0][mt][r]);
                float fg = sigf(acc[1][mt][r]);
                float gg = tanh_fast(acc[2][mt][r]);
                float og = sigf(acc[3][mt][r]);
                float c  = fg * c1[mt][r] + ig * gg;
                c1[mt][r] = c;
                sH1[(mt * 16 + r + hi8) * LST1 + col] = f2bf(og * tanh_fast(c));
            }
        }
        __syncthreads();        // h1_new visible

        // ---- decoder head: pred = h1 @ Wout^T + bout (4-way split dots) ----
        if (t >= OBS_LEN && tid < 384) {
            int q = tid & 3, pair = tid >> 2;
            int m = pair / 3, j = pair % 3;
            const unsigned short* hp = sH1 + m * LST1 + q * 64;
            const float* wp = sWout + j * HID + q * 64;
            float part = 0.0f;
#pragma unroll 8
            for (int k = 0; k < 64; ++k) part += bf2f(hp[k]) * wp[k];
            sP[pair][q] = part;
        }
        __syncthreads();
        if (t >= OBS_LEN && tid < 96) {
            int m = tid / 3, j = tid % 3;
            float s = bout[j] + sP[tid][0] + sP[tid][1] + sP[tid][2] + sP[tid][3];
            out[((size_t)(t - OBS_LEN) * B + rowbase + m) * 3 + j] = s;
            sH0[m * LST0 + HID + j] = f2bf(s);   // next-step x
        }
    }
}

extern "C" void kernel_launch(void* const* d_in, const int* in_sizes, int n_in,
                              void* d_out, int out_size, void* d_ws, size_t ws_size,
                              hipStream_t stream) {
    (void)n_in; (void)out_size; (void)ws_size;
    const float* obs  = (const float*)d_in[0];
    const float* Wih0 = (const float*)d_in[1];
    const float* Whh0 = (const float*)d_in[2];
    const float* bih0 = (const float*)d_in[3];
    const float* bhh0 = (const float*)d_in[4];
    const float* Wih1 = (const float*)d_in[5];
    const float* Whh1 = (const float*)d_in[6];
    const float* bih1 = (const float*)d_in[7];
    const float* bhh1 = (const float*)d_in[8];
    const float* Wout = (const float*)d_in[9];
    const float* bout = (const float*)d_in[10];

    int B = in_sizes[0] / (OBS_LEN * 3);              // 32768

    unsigned short* packed = (unsigned short*)d_ws;   // ~1.68 MB

    int packTotal = 2 * MAT9_ELEMS + MAT8_ELEMS;
    pack_kernel<<<(packTotal + 255) / 256, 256, 0, stream>>>(
        Whh0, Wih0, Wih1, Whh1, bih0, bhh0, bih1, bhh1, packed);

    lstm_kernel<<<B / MROWS, THREADS, 0, stream>>>(
        obs, Wout, bout, packed, (float*)d_out, B);
}
